// DisentangledAttention_20478404067604
// MI455X (gfx1250) — compile-verified, matrix operand order checked by reading, not run on hardware
//
#include <hip/hip_runtime.h>
#include <hip/hip_fp16.h>

// ---------------------------------------------------------------------------
// Disentangled attention, MI455X (gfx1250, wave32, WMMA).
// scores = (q.k)/8 + alpha_h*(pe_q.pe_k) + beta_h  folded into one GEMM by
// augmenting head-dim 64 -> 128:  Qa=[q/8, alpha*pe], Ka=[k, pe].
// f16 WMMA + fp32 accumulation; flash-style online softmax (no 512MB scores).
// GEMMs use 32x64 wave tiles with double-buffered fragments so loads overlap
// the v_wmma stream instead of serializing on s_wait_loadcnt 0.
// ---------------------------------------------------------------------------

typedef __attribute__((ext_vector_type(16))) _Float16 v16h;
typedef __attribute__((ext_vector_type(8)))  _Float16 v8h;
typedef __attribute__((ext_vector_type(8)))  float    v8f;

#define D_MODEL 1024
#define NHEADS  16
#define HDIM    64
#define SEQ     2048
#define BATCH   2
#define MROWS   (BATCH * SEQ)   // 4096
#define DAUG    128             // augmented head dim (64 content + 64 pos)

#define LDS_LDV 72
#define LDS_LDP 72

// ---- WMMA helper -----------------------------------------------------------
static __device__ __forceinline__ v8f wmma_f16(v16h a, v16h b, v8f c) {
    return __builtin_amdgcn_wmma_f32_16x16x32_f16(false, a, false, b, (short)0, c,
                                                  false, false);
}

// ---- 16-bit A-layout fragment load (also B-frag when given the N x K view) --
// ISA 7.12.2: lanes 0-15 hold row M=lane with K = {k0..k0+7, k0+16..k0+23};
// lanes 16-31 hold the same rows with K = {k0+8..k0+15, k0+24..k0+31}.
static __device__ __forceinline__ v16h load_frag(const _Float16* __restrict__ p,
                                                 int ld, int row0, int k0, int lane) {
    const int r  = lane & 15;
    const int h8 = (lane >> 4) << 3;   // 0 or 8
    const _Float16* base = p + (size_t)(row0 + r) * ld + k0 + h8;
    v8h lo = *(const v8h*)(base);
    v8h hi = *(const v8h*)(base + 16);
    return __builtin_shufflevector(lo, hi, 0,1,2,3,4,5,6,7,8,9,10,11,12,13,14,15);
}

// ---- fp32 -> f16 convert ---------------------------------------------------
__global__ void cvt_f32_f16_kernel(const float* __restrict__ in,
                                   _Float16* __restrict__ out, int n) {
    int i = blockIdx.x * blockDim.x + threadIdx.x;
    if (i < n) out[i] = (_Float16)in[i];
}

// ---- common 32x64 GEMM core: acc[mi][j] += A[m0+mi*16,:] . Wt[n0+j*16,:]^T --
// Double-buffered: next k-step's fragments are loaded into fresh registers
// while the 8 WMMAs for the current step issue.
static __device__ __forceinline__ void gemm_core_32x64(
        const _Float16* __restrict__ A, const _Float16* __restrict__ Wt,
        int m0, int n0, int lane, v8f acc[2][4]) {
    v16h a0 = load_frag(A, D_MODEL, m0,      0, lane);
    v16h a1 = load_frag(A, D_MODEL, m0 + 16, 0, lane);
    v16h b0 = load_frag(Wt, D_MODEL, n0,      0, lane);
    v16h b1 = load_frag(Wt, D_MODEL, n0 + 16, 0, lane);
    v16h b2 = load_frag(Wt, D_MODEL, n0 + 32, 0, lane);
    v16h b3 = load_frag(Wt, D_MODEL, n0 + 48, 0, lane);
#pragma unroll 4
    for (int k0 = 0; k0 < D_MODEL; k0 += 32) {
        v16h na0, na1, nb0, nb1, nb2, nb3;
        const int kn = k0 + 32;
        if (kn < D_MODEL) {
            na0 = load_frag(A, D_MODEL, m0,      kn, lane);
            na1 = load_frag(A, D_MODEL, m0 + 16, kn, lane);
            nb0 = load_frag(Wt, D_MODEL, n0,      kn, lane);
            nb1 = load_frag(Wt, D_MODEL, n0 + 16, kn, lane);
            nb2 = load_frag(Wt, D_MODEL, n0 + 32, kn, lane);
            nb3 = load_frag(Wt, D_MODEL, n0 + 48, kn, lane);
        }
        // 8 independent accumulation chains
        acc[0][0] = wmma_f16(a0, b0, acc[0][0]);
        acc[0][1] = wmma_f16(a0, b1, acc[0][1]);
        acc[0][2] = wmma_f16(a0, b2, acc[0][2]);
        acc[0][3] = wmma_f16(a0, b3, acc[0][3]);
        acc[1][0] = wmma_f16(a1, b0, acc[1][0]);
        acc[1][1] = wmma_f16(a1, b1, acc[1][1]);
        acc[1][2] = wmma_f16(a1, b2, acc[1][2]);
        acc[1][3] = wmma_f16(a1, b3, acc[1][3]);
        a0 = na0; a1 = na1;
        b0 = nb0; b1 = nb1; b2 = nb2; b3 = nb3;
    }
}

// ---- QKV projection: C = x @ W^T (+bias) * scale, head-split f16 output ----
// One wave computes a 32x64 tile. 2048 waves = 256 blocks x 8 waves.
__global__ void __launch_bounds__(256)
proj_qkv_kernel(const _Float16* __restrict__ A,    // MROWS x D_MODEL
                const _Float16* __restrict__ Wt,   // D_MODEL x D_MODEL (N x K)
                const float*    __restrict__ bias, // D_MODEL
                _Float16*       __restrict__ dst,  // [b,h,s,dstStride]
                int dstStride, float scale) {
    const int lane = threadIdx.x & 31;
    const int wid  = blockIdx.x * 8 + (threadIdx.x >> 5);
    const int m0   = (wid >> 4) << 5;   // 128 row tiles of 32
    const int n0   = (wid & 15) << 6;   // 16 col groups of 64

    v8f acc[2][4] = {};
    gemm_core_32x64(A, Wt, m0, n0, lane, acc);

    const int half = lane >> 4;
    const int nlo  = lane & 15;
#pragma unroll
    for (int mi = 0; mi < 2; ++mi)
#pragma unroll
        for (int j = 0; j < 4; ++j)
#pragma unroll
            for (int r = 0; r < 8; ++r) {
                int m = m0 + mi * 16 + r + half * 8;
                int n = n0 + j * 16 + nlo;
                int bb = m >> 11, s = m & (SEQ - 1);
                int hh = n >> 6,  d = n & (HDIM - 1);
                float v = (acc[mi][j][r] + bias[n]) * scale;
                dst[(((size_t)bb * NHEADS + hh) * SEQ + s) * dstStride + d] =
                    (_Float16)v;
            }
}

// ---- output projection: out = O @ Wo^T + bo (fp32 out) ---------------------
__global__ void __launch_bounds__(256)
proj_out_kernel(const _Float16* __restrict__ A,    // MROWS x D_MODEL (f16)
                const _Float16* __restrict__ Wt,   // D_MODEL x D_MODEL (N x K)
                const float*    __restrict__ bias,
                float*          __restrict__ out) {
    const int lane = threadIdx.x & 31;
    const int wid  = blockIdx.x * 8 + (threadIdx.x >> 5);
    const int m0   = (wid >> 4) << 5;
    const int n0   = (wid & 15) << 6;

    v8f acc[2][4] = {};
    gemm_core_32x64(A, Wt, m0, n0, lane, acc);

    const int half = lane >> 4;
    const int nlo  = lane & 15;
#pragma unroll
    for (int mi = 0; mi < 2; ++mi)
#pragma unroll
        for (int j = 0; j < 4; ++j)
#pragma unroll
            for (int r = 0; r < 8; ++r) {
                int m = m0 + mi * 16 + r + half * 8;
                int n = n0 + j * 16 + nlo;
                out[(size_t)m * D_MODEL + n] = acc[mi][j][r] + bias[n];
            }
}

// ---- fill positional halves of Qa / Ka -------------------------------------
__global__ void augment_kernel(const float* __restrict__ pe,
                               const float* __restrict__ alpha,
                               _Float16* __restrict__ Qa,
                               _Float16* __restrict__ Ka) {
    int i = blockIdx.x * blockDim.x + threadIdx.x;   // over B*H*S*64
    if (i >= BATCH * NHEADS * SEQ * HDIM) return;
    int d = i & (HDIM - 1);
    int s = (i >> 6) & (SEQ - 1);
    int h = (i >> 17) & (NHEADS - 1);
    float p = pe[s * HDIM + d];
    size_t base = ((size_t)(i >> 6)) * DAUG + HDIM + d;  // (b,h,s)*128 + 64 + d
    Qa[base] = (_Float16)(alpha[h] * p);
    Ka[base] = (_Float16)p;
}

// ---- fused flash attention over augmented dim 128 --------------------------
// Block = (b, h, 128 q rows) -> 8 waves, each wave owns 16 q rows.
__global__ void __launch_bounds__(256)
attn_kernel(const _Float16* __restrict__ Qa,   // [b,h,s,128]
            const _Float16* __restrict__ Ka,   // [b,h,s,128]
            const _Float16* __restrict__ V,    // [b,h,s,64]
            const int*      __restrict__ mask, // [b,s]
            const float*    __restrict__ beta, // [h]
            _Float16*       __restrict__ O) {  // [b,s,h,64] == MROWS x D_MODEL
    __shared__ __align__(16) _Float16 VT[HDIM * LDS_LDV];       // V^T tile [d][k]
    __shared__ __align__(16) _Float16 PS[8 * 16 * LDS_LDP];     // per-wave P

    const int lane = threadIdx.x & 31;
    const int widx = threadIdx.x >> 5;
    const int bh   = blockIdx.x >> 4;           // b*16 + h
    const int b    = bh >> 4;
    const int h    = bh & 15;
    const int q0   = (blockIdx.x & 15) * 128 + widx * 16;

    const _Float16* Qh = Qa + (size_t)bh * SEQ * DAUG;
    const _Float16* Kh = Ka + (size_t)bh * SEQ * DAUG;
    const _Float16* Vh = V  + (size_t)bh * SEQ * HDIM;
    const float bet = beta[h];

    v16h qf[4];
#pragma unroll
    for (int kq = 0; kq < 4; ++kq) qf[kq] = load_frag(Qh, DAUG, q0, kq * 32, lane);

    v8f oacc[4] = {};
    float rmax[8], rsum[8];
#pragma unroll
    for (int r = 0; r < 8; ++r) { rmax[r] = -3.0e38f; rsum[r] = 0.f; }

    const int half = lane >> 4;
    const int nlo  = lane & 15;
    _Float16* myP = &PS[widx * 16 * LDS_LDP];

    for (int k0 = 0; k0 < SEQ; k0 += 64) {
        // ---- cooperative V^T staging (256 threads, 64x64 f16 tile) ----
        {
            int kk = threadIdx.x & 63;
            int d0 = (threadIdx.x >> 6) * 16;
            const _Float16* vp = Vh + (size_t)(k0 + kk) * HDIM + d0;
            v8h a = *(const v8h*)vp;
            v8h c = *(const v8h*)(vp + 8);
#pragma unroll
            for (int i = 0; i < 8; ++i) {
                VT[(d0 + i)     * LDS_LDV + kk] = a[i];
                VT[(d0 + 8 + i) * LDS_LDV + kk] = c[i];
            }
            if (k0 + 64 < SEQ)  // pull next V tile toward L2/L0
                __builtin_prefetch(Vh + (size_t)(k0 + 64 + kk) * HDIM + d0, 0, 1);
        }
        __syncthreads();

        // ---- scores S = Qa.Ka^T (16x64, K=128); kq-outer => 4 independent
        //      WMMA chains per step, 4 K-frag loads clustered per step ----
        v8f sc[4] = {};
#pragma unroll
        for (int kq = 0; kq < 4; ++kq) {
            v16h kb[4];
#pragma unroll
            for (int j = 0; j < 4; ++j)
                kb[j] = load_frag(Kh, DAUG, k0 + j * 16, kq * 32, lane);
#pragma unroll
            for (int j = 0; j < 4; ++j)
                sc[j] = wmma_f16(qf[kq], kb[j], sc[j]);
        }
#pragma unroll
        for (int j = 0; j < 4; ++j) {
            int mv = mask[b * SEQ + k0 + j * 16 + nlo];
#pragma unroll
            for (int r = 0; r < 8; ++r)
                sc[j][r] = (mv == 0) ? -3.0e38f : (sc[j][r] + bet);
        }

        // ---- online softmax (rows live across 16 lanes of each half) ----
#pragma unroll
        for (int r = 0; r < 8; ++r) {
            float m = fmaxf(fmaxf(sc[0][r], sc[1][r]), fmaxf(sc[2][r], sc[3][r]));
#pragma unroll
            for (int off = 1; off < 16; off <<= 1)
                m = fmaxf(m, __shfl_xor(m, off, 32));
            float nm   = fmaxf(rmax[r], m);
            float corr = __expf(rmax[r] - nm);
            rmax[r] = nm;
            rsum[r] *= corr;
#pragma unroll
            for (int j = 0; j < 4; ++j) oacc[j][r] *= corr;
            float ls = 0.f;
#pragma unroll
            for (int j = 0; j < 4; ++j) {
                float p = __expf(sc[j][r] - nm);
                sc[j][r] = p;
                ls += p;
            }
#pragma unroll
            for (int off = 1; off < 16; off <<= 1)
                ls += __shfl_xor(ls, off, 32);
            rsum[r] += ls;
        }

        // ---- stage P (C layout -> row-major LDS) ----
#pragma unroll
        for (int j = 0; j < 4; ++j)
#pragma unroll
            for (int r = 0; r < 8; ++r)
                myP[(r + half * 8) * LDS_LDP + j * 16 + nlo] = (_Float16)sc[j][r];

        // ---- P.V : A-frags from P stage, B-frags from V^T stage ----
#pragma unroll
        for (int kk = 0; kk < 2; ++kk) {
            v16h pa = load_frag(myP, LDS_LDP, 0, kk * 32, lane);
            v16h vb[4];
#pragma unroll
            for (int j2 = 0; j2 < 4; ++j2)
                vb[j2] = load_frag(VT, LDS_LDV, j2 * 16, kk * 32, lane);
#pragma unroll
            for (int j2 = 0; j2 < 4; ++j2)
                oacc[j2] = wmma_f16(pa, vb[j2], oacc[j2]);
        }
        __syncthreads();   // protect VT before next staging round
    }

    // ---- normalize and store O in [b,s,h,d] (row-major 4096 x 1024) ----
#pragma unroll
    for (int r = 0; r < 8; ++r) {
        float inv = 1.f / fmaxf(rsum[r], 1e-20f);
        int s = q0 + r + half * 8;
#pragma unroll
        for (int j2 = 0; j2 < 4; ++j2) {
            int d = j2 * 16 + nlo;
            O[(((size_t)b * SEQ + s) * NHEADS + h) * HDIM + d] =
                (_Float16)(oacc[j2][r] * inv);
        }
    }
}

// ---------------------------------------------------------------------------
extern "C" void kernel_launch(void* const* d_in, const int* in_sizes, int n_in,
                              void* d_out, int out_size, void* d_ws, size_t ws_size,
                              hipStream_t stream) {
    const float* x     = (const float*)d_in[0];
    const int*   mask  = (const int*)d_in[1];
    const float* Wq    = (const float*)d_in[2];
    const float* bq    = (const float*)d_in[3];
    const float* Wk    = (const float*)d_in[4];
    const float* bk    = (const float*)d_in[5];
    const float* Wv    = (const float*)d_in[6];
    const float* bv    = (const float*)d_in[7];
    const float* Wo    = (const float*)d_in[8];
    const float* bo    = (const float*)d_in[9];
    const float* pe    = (const float*)d_in[10];
    const float* alpha = (const float*)d_in[11];
    const float* beta  = (const float*)d_in[12];
    float* out = (float*)d_out;

    // workspace carve-up (64 MB total)
    char* ws = (char*)d_ws;
    _Float16* xh  = (_Float16*)ws; ws += (size_t)MROWS * D_MODEL * 2;        // 8 MB
    _Float16* Wqh = (_Float16*)ws; ws += (size_t)D_MODEL * D_MODEL * 2;      // 2 MB
    _Float16* Wkh = (_Float16*)ws; ws += (size_t)D_MODEL * D_MODEL * 2;
    _Float16* Wvh = (_Float16*)ws; ws += (size_t)D_MODEL * D_MODEL * 2;
    _Float16* Woh = (_Float16*)ws; ws += (size_t)D_MODEL * D_MODEL * 2;
    _Float16* Qa  = (_Float16*)ws; ws += (size_t)BATCH * NHEADS * SEQ * DAUG * 2; // 16 MB
    _Float16* Kaa = (_Float16*)ws; ws += (size_t)BATCH * NHEADS * SEQ * DAUG * 2; // 16 MB
    _Float16* Vb  = (_Float16*)ws; ws += (size_t)BATCH * NHEADS * SEQ * HDIM * 2; //  8 MB
    _Float16* Oh  = (_Float16*)ws; ws += (size_t)MROWS * D_MODEL * 2;             //  8 MB

    int n;
    n = MROWS * D_MODEL;
    cvt_f32_f16_kernel<<<(n + 255) / 256, 256, 0, stream>>>(x, xh, n);
    n = D_MODEL * D_MODEL;
    cvt_f32_f16_kernel<<<(n + 255) / 256, 256, 0, stream>>>(Wq, Wqh, n);
    cvt_f32_f16_kernel<<<(n + 255) / 256, 256, 0, stream>>>(Wk, Wkh, n);
    cvt_f32_f16_kernel<<<(n + 255) / 256, 256, 0, stream>>>(Wv, Wvh, n);
    cvt_f32_f16_kernel<<<(n + 255) / 256, 256, 0, stream>>>(Wo, Woh, n);

    // q scaled by 1/sqrt(HDIM)=1/8 folded into Qa content half
    proj_qkv_kernel<<<256, 256, 0, stream>>>(xh, Wqh, bq, Qa, DAUG, 0.125f);
    proj_qkv_kernel<<<256, 256, 0, stream>>>(xh, Wkh, bk, Kaa, DAUG, 1.0f);
    proj_qkv_kernel<<<256, 256, 0, stream>>>(xh, Wvh, bv, Vb, HDIM, 1.0f);

    n = BATCH * NHEADS * SEQ * HDIM;
    augment_kernel<<<(n + 255) / 256, 256, 0, stream>>>(pe, alpha, Qa, Kaa);

    attn_kernel<<<512, 256, 0, stream>>>(Qa, Kaa, Vb, mask, beta, Oh);

    proj_out_kernel<<<256, 256, 0, stream>>>(Oh, Woh, bo, out);
}
